// EllEssTeeEmm_62405874811230
// MI455X (gfx1250) — matrix operand order repaired and checked by reading, not compiled, not run on hardware
//
#include <hip/hip_runtime.h>
#include <hip/hip_bf16.h>
#include <math.h>

// ---------------------------------------------------------------------------
// Types for CDNA5 WMMA (wave32): 16x16x32 bf16 -> f32
// ---------------------------------------------------------------------------
typedef __bf16 bf16_t;
typedef bf16_t v16bf __attribute__((ext_vector_type(16)));
typedef bf16_t v8bf  __attribute__((ext_vector_type(8)));
typedef float  v8f   __attribute__((ext_vector_type(8)));
typedef unsigned int u32x4 __attribute__((ext_vector_type(4)));
typedef int          i32x4 __attribute__((ext_vector_type(4)));
typedef int          i32x8 __attribute__((ext_vector_type(8)));

#if defined(__has_builtin)
#if __has_builtin(__builtin_amdgcn_tensor_load_to_lds)
#define ELLESS_HAVE_TDM 1
#endif
#endif

union V16U { v16bf v; v8bf h[2]; };

static __device__ __forceinline__ v8f wmma_bf16(v16bf a, v16bf b, v8f c) {
    return __builtin_amdgcn_wmma_f32_16x16x32_bf16(
        false, a, false, b, (short)0, c, /*reuse_a=*/false, /*reuse_b=*/false);
}
// Same instruction as the previous one with identical A operand -> reuse_a legal
static __device__ __forceinline__ v8f wmma_bf16_ra(v16bf a, v16bf b, v8f c) {
    return __builtin_amdgcn_wmma_f32_16x16x32_bf16(
        false, a, false, b, (short)0, c, /*reuse_a=*/true, /*reuse_b=*/false);
}

// A fragment (16x32 bf16). lane<16: row=lane, K [kk..kk+16);
// lane>=16: row=lane-16, K [kk+16..kk+32).
static __device__ __forceinline__ v16bf load_a_row(const bf16_t* rowp, int kk, int lane) {
    const int klo = (lane >> 4) * 16;
    V16U u;
    u.h[0] = *(const v8bf*)(rowp + kk + klo);
    u.h[1] = *(const v8bf*)(rowp + kk + 8 + klo);
    return u.v;
}
// B fragment (32x16 bf16): lane holds col = lane&15, 16 contiguous K values
// starting at kk + (lane>>4)*16.  colp = pointer to column (= W row).
static __device__ __forceinline__ v16bf load_b_col(const bf16_t* colp, int kk, int lane) {
    return *(const v16bf*)(colp + kk + ((lane >> 4) << 4));
}

static __device__ __forceinline__ float sigmoidf_fast(float x) {
    return 1.0f / (1.0f + __expf(-x));
}

// ---------------------------------------------------------------------------
// fp32 -> bf16 conversion
// ---------------------------------------------------------------------------
__global__ void EllEss_cvt_bf16_kernel(const float* __restrict__ src,
                                       bf16_t* __restrict__ dst, long n) {
    long i = (long)blockIdx.x * blockDim.x + threadIdx.x;
    if (i < n) dst[i] = (bf16_t)src[i];
}

// ---------------------------------------------------------------------------
// xg = X @ W_ih^T + b_ih + b_hh   (one direction)
// X: [B=32, T, Din] bf16.  W_ih: [1024, Din] bf16.
// xg out layout: [T][gate_col 0..1023][batch 0..31] bf16 so the scan can read
// one v8bf per gate per lane, and the epilogue here stores one v8bf per tile.
// Grid: (T*B/16, 2), block 256 (8 waves); wave computes a 16x64 strip
// (4 N-tiles) so the A fragment is loaded once per K-step and reused 4x.
// ---------------------------------------------------------------------------
__global__ __launch_bounds__(256)
void EllEss_gemm_xg_kernel(const bf16_t* __restrict__ X,
                           const bf16_t* __restrict__ Wih,
                           const float* __restrict__ b_ih,
                           const float* __restrict__ b_hh,
                           bf16_t* __restrict__ xg,
                           int T, int Din) {
    const int lane = threadIdx.x & 31;
    const int wave = threadIdx.x >> 5;
    const int mt  = blockIdx.x;                        // M tile: rows r = t*32+b
    const int nt0 = (blockIdx.y * 8 + wave) * 4;       // first of 4 N tiles

    const int r = mt * 16 + (lane & 15);
    const int b = r & 31;
    const int t = r >> 5;
    const bf16_t* arow = X + ((size_t)b * T + t) * Din;
    const bf16_t* bcol = Wih + (size_t)(nt0 * 16 + (lane & 15)) * Din;
    const size_t bstride = (size_t)16 * Din;

    v8f acc0 = {}, acc1 = {}, acc2 = {}, acc3 = {};
    for (int kk = 0; kk < Din; kk += 32) {
        v16bf a = load_a_row(arow, kk, lane);
        acc0 = wmma_bf16(a, load_b_col(bcol,               kk, lane), acc0);
        acc1 = wmma_bf16_ra(a, load_b_col(bcol + bstride,     kk, lane), acc1);
        acc2 = wmma_bf16_ra(a, load_b_col(bcol + 2 * bstride, kk, lane), acc2);
        acc3 = wmma_bf16_ra(a, load_b_col(bcol + 3 * bstride, kk, lane), acc3);
    }

    const int b0   = (mt & 1) * 16 + ((lane >> 4) << 3);  // batch base (0..31)
    const int trow = mt >> 1;
    auto epi = [&](v8f acc, int ntile) {
        const int col = ntile * 16 + (lane & 15);
        const float bias = b_ih[col] + b_hh[col];
        v8bf out;
#pragma unroll
        for (int i = 0; i < 8; ++i) out[i] = (bf16_t)(acc[i] + bias);
        *(v8bf*)(xg + ((size_t)trow * 1024 + col) * 32 + b0) = out;
    };
    epi(acc0, nt0); epi(acc1, nt0 + 1); epi(acc2, nt0 + 2); epi(acc3, nt0 + 3);
}

// ---------------------------------------------------------------------------
// Zero h double-buffers and the per-direction step counters.
// ---------------------------------------------------------------------------
__global__ void EllEss_scan_init_kernel(bf16_t* __restrict__ hbuf,
                                        int* __restrict__ cnt) {
    int i = blockIdx.x * blockDim.x + threadIdx.x;
    for (int k = i; k < 2 * 2 * 32 * 256; k += blockDim.x * gridDim.x)
        hbuf[k] = (bf16_t)0.0f;
    if (i < 2) cnt[i] = 0;
}

// ---------------------------------------------------------------------------
// Recurrent scan, one layer, both directions.
// Grid: (8 WGs per direction, 2 directions), block 128 (4 waves).
// WG `wg` owns hidden units [wg*32, wg*32+32) -> 128 gate columns.
// Its W_hh slice (128 x 256 bf16 = 64KB) is staged into LDS by the Tensor
// Data Mover (4 descriptors, one 32-row tile per gate), then 1024 serial
// steps each run a K=256 WMMA chain per gate, fused LSTM pointwise update,
// double-buffered h in L2, and a device-scope atomic spin barrier.
// ---------------------------------------------------------------------------
#define NWG 8
__global__ __launch_bounds__(128)
void EllEss_scan_kernel(const bf16_t* __restrict__ xg,    // [2][T][1024][32]
                        const bf16_t* __restrict__ Whh,   // [2][1024][256]
                        bf16_t* __restrict__ hbuf,        // [2 dirs][2 par][32][256]
                        float* __restrict__ outf,         // [32][T][512] or null
                        bf16_t* __restrict__ nextbf,      // [32][T][512] or null
                        int* __restrict__ cnt,            // [2]
                        int T) {
    const int dir   = blockIdx.y;
    const int wg    = blockIdx.x;            // 0..7
    const int lane  = threadIdx.x & 31;
    const int wave  = threadIdx.x >> 5;      // 0..3
    const int Mtile = wave & 1;
    const int jt    = wave >> 1;
    const bool rev  = (dir == 1);

    __shared__ alignas(32) bf16_t ldsW[128 * 256];  // 64 KB of 320 KB/WGP

#if defined(ELLESS_HAVE_TDM)
    // Stage W_hh slice with the Tensor Data Mover: one 2D tile per gate,
    // 32 rows x 256 bf16 (512B) -> LDS rows nloc = gate*32 + jj.
    if (wave == 0) {
        const unsigned lds_base = (unsigned)(uintptr_t)(&ldsW[0]); // low 32b = LDS offset
        for (int g = 0; g < 4; ++g) {
            const unsigned long long gaddr = (unsigned long long)(uintptr_t)(
                Whh + ((size_t)dir * 1024 + g * 256 + wg * 32) * 256);
            u32x4 g0;
            g0[0] = 1u;                                   // count=1, user descriptor
            g0[1] = lds_base + (unsigned)(g * 32 * 256 * 2);
            g0[2] = (unsigned)(gaddr & 0xFFFFFFFFu);      // global_addr[31:0]
            g0[3] = (unsigned)((gaddr >> 32) & 0x1FFFFFFu) | (2u << 30); // addr[56:32] | type=2
            i32x8 g1;
            g1[0] = (int)(1u << 16);      // workgroup_mask=0, data_size=1 (2 bytes)
            g1[1] = (int)(256u << 16);    // tensor_dim0 = 256 (lo16)
            g1[2] = (int)(1024u << 16);   // tensor_dim0 hi=0 | tensor_dim1 lo16 = 1024
            g1[3] = (int)(256u << 16);    // tensor_dim1 hi=0 | tile_dim0 = 256
            g1[4] = 32;                   // tile_dim1 = 32, tile_dim2 = 0
            g1[5] = 256;                  // tensor_dim0_stride lo32 = 256
            g1[6] = 0;
            g1[7] = 0;
            i32x4 gz = {0, 0, 0, 0};
#if defined(__clang_major__) && (__clang_major__ >= 23)
            i32x8 gz8 = {0, 0, 0, 0, 0, 0, 0, 0};
            __builtin_amdgcn_tensor_load_to_lds(g0, g1, gz, gz, gz8, 0);
#else
            __builtin_amdgcn_tensor_load_to_lds(g0, g1, gz, gz, 0);
#endif
        }
        __builtin_amdgcn_s_wait_tensorcnt(0);
    }
#else
    for (int i = threadIdx.x; i < 128 * 16; i += 128) {
        const int nloc = i >> 4, seg = i & 15;
        const int gate = nloc >> 5, jj = nloc & 31;
        const int ng = gate * 256 + wg * 32 + jj;
        *(v16bf*)(ldsW + (size_t)nloc * 256 + seg * 16) =
            *(const v16bf*)(Whh + ((size_t)dir * 1024 + ng) * 256 + seg * 16);
    }
#endif
    __syncthreads();

    const int col     = lane & 15;
    const int jwg     = jt * 16 + col;        // 0..31 within WG slice
    const int jglob   = wg * 32 + jwg;        // 0..255 hidden index
    const int rowbase = Mtile * 16 + ((lane >> 4) << 3);   // batch base 0..31
    const int kb_off  = (lane >> 4) << 4;

    v8f c = {};                               // cell state, 8 batches per lane

    for (int t = 0; t < T; ++t) {
        const int te  = rev ? (T - 1 - t) : t;
        const int par = t & 1;

        const bf16_t* xgt = xg + ((size_t)dir * T + te) * (1024 * 32);
        __builtin_prefetch(xgt + (size_t)jglob * 32 + rowbase, 0, 1);

        const bf16_t* hprev = hbuf + ((size_t)dir * 2 + par) * (32 * 256);
        const bf16_t* harow = hprev + (size_t)(Mtile * 16 + col) * 256;

        v8f ai = {}, af = {}, ag = {}, ao = {};
#pragma unroll
        for (int kk = 0; kk < 256; kk += 32) {
            v16bf a = load_a_row(harow, kk, lane);
            const int kb = kk + kb_off;
            v16bf b0 = *(const v16bf*)(ldsW + (size_t)(0 * 32 + jwg) * 256 + kb);
            v16bf b1 = *(const v16bf*)(ldsW + (size_t)(1 * 32 + jwg) * 256 + kb);
            v16bf b2 = *(const v16bf*)(ldsW + (size_t)(2 * 32 + jwg) * 256 + kb);
            v16bf b3 = *(const v16bf*)(ldsW + (size_t)(3 * 32 + jwg) * 256 + kb);
            ai = wmma_bf16(a, b0, ai);
            af = wmma_bf16_ra(a, b1, af);
            ag = wmma_bf16_ra(a, b2, ag);
            ao = wmma_bf16_ra(a, b3, ao);
        }

        // One coalesced 16B load per gate: xg[dir][te][gate*256+j][b0..b0+8)
        v8bf xvi = *(const v8bf*)(xgt + ((size_t)(0 * 256 + jglob)) * 32 + rowbase);
        v8bf xvf = *(const v8bf*)(xgt + ((size_t)(1 * 256 + jglob)) * 32 + rowbase);
        v8bf xvg = *(const v8bf*)(xgt + ((size_t)(2 * 256 + jglob)) * 32 + rowbase);
        v8bf xvo = *(const v8bf*)(xgt + ((size_t)(3 * 256 + jglob)) * 32 + rowbase);

        bf16_t* hnext = hbuf + ((size_t)dir * 2 + (1 - par)) * (32 * 256);
#pragma unroll
        for (int i = 0; i < 8; ++i) {
            const int bidx = rowbase + i;
            const float gi = ai[i] + (float)xvi[i];
            const float gf = af[i] + (float)xvf[i];
            const float gg = ag[i] + (float)xvg[i];
            const float go = ao[i] + (float)xvo[i];
            const float sI = sigmoidf_fast(gi);
            const float sF = sigmoidf_fast(gf);
            const float sO = sigmoidf_fast(go);
            const float cn = sF * c[i] + sI * tanhf(gg);
            c[i] = cn;
            const float h = sO * tanhf(cn);
            const bf16_t hb = (bf16_t)h;
            hnext[(size_t)bidx * 256 + jglob] = hb;
            const size_t oidx =
                ((size_t)bidx * T + te) * 512 + (size_t)dir * 256 + jglob;
            if (outf)   outf[oidx]   = h;
            if (nextbf) nextbf[oidx] = hb;
        }

        // Device-scope step barrier across the 8 WGs of this direction.
        __threadfence();
        __syncthreads();
        if (threadIdx.x == 0) {
            __hip_atomic_fetch_add(&cnt[dir], 1, __ATOMIC_ACQ_REL,
                                   __HIP_MEMORY_SCOPE_AGENT);
            const int target = (t + 1) * NWG;
            while (__hip_atomic_load(&cnt[dir], __ATOMIC_ACQUIRE,
                                     __HIP_MEMORY_SCOPE_AGENT) < target)
                __builtin_amdgcn_s_sleep(1);
        }
        __syncthreads();
        __builtin_amdgcn_fence(__ATOMIC_ACQUIRE, "agent");
    }
}

// ---------------------------------------------------------------------------
// Host-side orchestration
// ---------------------------------------------------------------------------
extern "C" void kernel_launch(void* const* d_in, const int* in_sizes, int n_in,
                              void* d_out, int out_size, void* d_ws, size_t ws_size,
                              hipStream_t stream) {
    (void)in_sizes; (void)n_in; (void)out_size; (void)ws_size;
    const int B = 32, T = 1024, IN = 256, H = 256;

    const float* x = (const float*)d_in[0];
    const float* Wih_f0 = (const float*)d_in[1];
    const float* Whh_f0 = (const float*)d_in[2];
    const float* bih_f0 = (const float*)d_in[3];
    const float* bhh_f0 = (const float*)d_in[4];
    const float* Wih_r0 = (const float*)d_in[5];
    const float* Whh_r0 = (const float*)d_in[6];
    const float* bih_r0 = (const float*)d_in[7];
    const float* bhh_r0 = (const float*)d_in[8];
    const float* Wih_f1 = (const float*)d_in[9];
    const float* Whh_f1 = (const float*)d_in[10];
    const float* bih_f1 = (const float*)d_in[11];
    const float* bhh_f1 = (const float*)d_in[12];
    const float* Wih_r1 = (const float*)d_in[13];
    const float* Whh_r1 = (const float*)d_in[14];
    const float* bih_r1 = (const float*)d_in[15];
    const float* bhh_r1 = (const float*)d_in[16];

    char* base = (char*)d_ws;
    size_t off = 0;
    auto take = [&](size_t bytes) -> char* {
        char* p = base + off;
        off += (bytes + 255) & ~(size_t)255;
        return p;
    };
    bf16_t* xbf0 = (bf16_t*)take((size_t)B * T * IN * 2);        // 16 MB
    bf16_t* xbf1 = (bf16_t*)take((size_t)B * T * 2 * H * 2);     // 32 MB
    bf16_t* xg   = (bf16_t*)take((size_t)2 * T * B * 4 * H * 2); // 128 MB
    bf16_t* wih0 = (bf16_t*)take((size_t)2 * 4 * H * IN * 2);
    bf16_t* wih1 = (bf16_t*)take((size_t)2 * 4 * H * 2 * H * 2);
    bf16_t* whh0 = (bf16_t*)take((size_t)2 * 4 * H * H * 2);
    bf16_t* whh1 = (bf16_t*)take((size_t)2 * 4 * H * H * 2);
    bf16_t* hbuf = (bf16_t*)take((size_t)2 * 2 * 32 * H * 2);
    int*    cnt  = (int*)take(256);

    auto cvt = [&](const float* s, bf16_t* d, long n) {
        EllEss_cvt_bf16_kernel<<<(unsigned)((n + 255) / 256), 256, 0, stream>>>(s, d, n);
    };

    cvt(x, xbf0, (long)B * T * IN);
    cvt(Wih_f0, wih0,                 (long)4 * H * IN);
    cvt(Wih_r0, wih0 + 4 * H * IN,    (long)4 * H * IN);
    cvt(Whh_f0, whh0,                 (long)4 * H * H);
    cvt(Whh_r0, whh0 + 4 * H * H,     (long)4 * H * H);
    cvt(Wih_f1, wih1,                 (long)4 * H * 2 * H);
    cvt(Wih_r1, wih1 + 4 * H * 2 * H, (long)4 * H * 2 * H);
    cvt(Whh_f1, whh1,                 (long)4 * H * H);
    cvt(Whh_r1, whh1 + 4 * H * H,     (long)4 * H * H);

    const size_t xg_dir = (size_t)T * B * 4 * H;
    const dim3 ggrid(T * B / 16, 2);

    // ---- Layer 0 ----
    EllEss_gemm_xg_kernel<<<ggrid, 256, 0, stream>>>(
        xbf0, wih0, bih_f0, bhh_f0, xg, T, IN);
    EllEss_gemm_xg_kernel<<<ggrid, 256, 0, stream>>>(
        xbf0, wih0 + 4 * H * IN, bih_r0, bhh_r0, xg + xg_dir, T, IN);
    EllEss_scan_init_kernel<<<8, 256, 0, stream>>>(hbuf, cnt);
    EllEss_scan_kernel<<<dim3(NWG, 2), 128, 0, stream>>>(
        xg, whh0, hbuf, nullptr, xbf1, cnt, T);

    // ---- Layer 1 ----
    EllEss_gemm_xg_kernel<<<ggrid, 256, 0, stream>>>(
        xbf1, wih1, bih_f1, bhh_f1, xg, T, 2 * H);
    EllEss_gemm_xg_kernel<<<ggrid, 256, 0, stream>>>(
        xbf1, wih1 + 4 * H * 2 * H, bih_r1, bhh_r1, xg + xg_dir, T, 2 * H);
    EllEss_scan_init_kernel<<<8, 256, 0, stream>>>(hbuf, cnt);
    EllEss_scan_kernel<<<dim3(NWG, 2), 128, 0, stream>>>(
        xg, whh1, hbuf, (float*)d_out, nullptr, cnt, T);
}
